// CrossAttention_43224550867585
// MI455X (gfx1250) — matrix operand order repaired and checked by reading, not compiled, not run on hardware
//
#include <hip/hip_runtime.h>
#include <hip/hip_bf16.h>

typedef __bf16 bf16_t;
typedef __attribute__((ext_vector_type(16))) __bf16 v16bf;
typedef __attribute__((ext_vector_type(8)))  __bf16 v8bf;
typedef __attribute__((ext_vector_type(4)))  __bf16 v4bf;
typedef __attribute__((ext_vector_type(8)))  float  v8f;
typedef __attribute__((ext_vector_type(4)))  unsigned uint4v;
typedef __attribute__((ext_vector_type(8)))  unsigned uint8v;

#define B_BATCH 2
#define LQ 2048
#define LK 2048
#define DM 1024
#define NH 16
#define DH 64
#define QK_SCALE 0.125f     /* 1/sqrt(64), folded into normalized Q */
#define EPS_SM 0.1f

struct BFPair { v8bf lo, hi; };
static __device__ __forceinline__ v16bf mk16(v8bf lo, v8bf hi) {
  BFPair t; t.lo = lo; t.hi = hi;
  return __builtin_bit_cast(v16bf, t);
}
static __device__ __forceinline__ v8f wmma_bf16(v16bf a, v16bf b, v8f c) {
  return __builtin_amdgcn_wmma_f32_16x16x32_bf16(false, a, false, b, (short)0, c,
                                                 false, false);
}

/* Async 16-byte global->LDS copy (GVS mode), tracked by ASYNCcnt. */
static __device__ __forceinline__ void async_b128(unsigned ldsoff,
                                                  const void* base,
                                                  unsigned goff) {
  asm volatile("global_load_async_to_lds_b128 %0, %1, %2"
               :: "v"(ldsoff), "v"(goff), "s"(base) : "memory");
}
#define LDSOFF(p) ((unsigned)(unsigned long long)(p))

/* TDM 2-D tile load, global -> LDS, tracked by TENSORcnt.
   tensor_w/h in elements (2B), stride_w in elements, tile_w x tile_h tile.
   pad_interval: pad after 2^interval * 8 bytes; pad_amount: (n+1) DWORDs. */
static __device__ __forceinline__ void tdm_load_2d(
    unsigned ldsoff, const void* gaddr,
    unsigned tensor_w, unsigned tensor_h, unsigned stride_w,
    unsigned tile_w, unsigned tile_h,
    unsigned pad_interval, unsigned pad_amount)
{
  unsigned long long ga = (unsigned long long)gaddr;
  uint4v g0;
  g0[0] = 1u;                                    /* count=1, user desc */
  g0[1] = ldsoff;                                /* lds_addr            */
  g0[2] = (unsigned)ga;                          /* global_addr[31:0]   */
  g0[3] = (unsigned)(ga >> 32) | (2u << 30);     /* addr[56:32] | type=2 */
  uint8v g1;
  g1[0] = (1u << 16) | (1u << 20)                /* data_size=2B, pad_en */
        | (pad_interval << 22) | (pad_amount << 25);
  g1[1] = tensor_w << 16;                        /* tensor_dim0[15:0]   */
  g1[2] = (tensor_w >> 16) | (tensor_h << 16);   /* td0[31:16]|td1[15:0] */
  g1[3] = (tensor_h >> 16) | (tile_w << 16);     /* td1[31:16]|tile_dim0 */
  g1[4] = tile_h;                                /* tile_dim1, tile_dim2=0 */
  g1[5] = stride_w;                              /* tensor_dim0_stride[31:0] */
  g1[6] = 0u;
  g1[7] = 0u;
  asm volatile("tensor_load_to_lds %0, %1" :: "s"(g0), "s"(g1) : "memory");
}

/* ---- fp32 -> bf16 elementwise (activations) ---- */
__global__ __launch_bounds__(256) void conv_kernel(const float* __restrict__ X,
                                                   bf16_t* __restrict__ Y) {
  int idx = (blockIdx.x * 256 + threadIdx.x) * 4;
  const float4 f = *reinterpret_cast<const float4*>(&X[idx]);
  v4bf h; h[0]=(bf16_t)f.x; h[1]=(bf16_t)f.y; h[2]=(bf16_t)f.z; h[3]=(bf16_t)f.w;
  *reinterpret_cast<v4bf*>(&Y[idx]) = h;
}

/* ---- W (N x K, fp32) -> WT (K x N, bf16) ---- */
__global__ __launch_bounds__(256) void wt_kernel(const float* __restrict__ W,
                                                 bf16_t* __restrict__ WT) {
  int idx = blockIdx.x * 256 + threadIdx.x;
  int n = idx >> 10, k = idx & (DM - 1);
  WT[k * DM + n] = (bf16_t)W[idx];
}

/* ---- C = A(bf16, MxK) @ WT(bf16, KxN) + bias, fp32 out ----
   128x128 tile per 256-thread WG (8 waves), wave = 32x64.
   Double-buffered TDM (tensor_load_to_lds) staging pipelined with WMMA. */
__global__ __launch_bounds__(256) void gemm_kernel(const bf16_t* __restrict__ A,
    const bf16_t* __restrict__ BT, const float* __restrict__ bias,
    float* __restrict__ C, int M, int N, int K)
{
  __shared__ bf16_t As[2][128][40];   /* 32-elem rows + 16B pad */
  __shared__ bf16_t Bs[2][32][136];   /* 128-elem rows + 16B pad */
  const int tid = threadIdx.x;
  const int lane = tid & 31, wid = tid >> 5;
  const int wm = (wid & 3) * 32;
  const int wn = (wid >> 2) * 64;
  const int m0 = blockIdx.y * 128;
  const int n0 = blockIdx.x * 128;
  const int rhalf = lane >> 4;
  const int rlow  = lane & 15;

  __builtin_prefetch(&bias[n0 + tid], 0, 0);

  v8f acc[2][4];
  #pragma unroll
  for (int i = 0; i < 2; i++)
    #pragma unroll
    for (int j = 0; j < 4; j++) acc[i][j] = 0.f;

  auto stage = [&](int buf, int k0) {
    if (wid == 0) {
      /* A tile: 128 rows x 32 elems (64B rows -> pad_interval 3) */
      tdm_load_2d(LDSOFF(&As[buf][0][0]), A + (size_t)m0 * K + k0,
                  (unsigned)K, (unsigned)M, (unsigned)K, 32u, 128u, 3u, 3u);
      /* B tile: 32 rows x 128 elems (256B rows -> pad_interval 5) */
      tdm_load_2d(LDSOFF(&Bs[buf][0][0]), BT + (size_t)k0 * N + n0,
                  (unsigned)N, (unsigned)K, (unsigned)N, 128u, 32u, 5u, 3u);
    }
  };

  stage(0, 0);
  const int nk = K / 32;
  for (int ki = 0; ki < nk; ki++) {
    const int buf = ki & 1;
    if (ki + 1 < nk) {
      stage(buf ^ 1, (ki + 1) * 32);
      __builtin_amdgcn_s_wait_tensorcnt(2);  /* tile ki's 2 TDMs done */
    } else {
      __builtin_amdgcn_s_wait_tensorcnt(0);
    }
    __syncthreads();

    v16bf af[2], bfm[4];
    #pragma unroll
    for (int mi = 0; mi < 2; mi++) {
      int r = wm + mi * 16 + rlow;
      int cb = rhalf * 8;
      af[mi] = mk16(*reinterpret_cast<v8bf*>(&As[buf][r][cb]),
                    *reinterpret_cast<v8bf*>(&As[buf][r][cb + 16]));
    }
    #pragma unroll
    for (int ni = 0; ni < 4; ni++) {
      int c = wn + ni * 16;
      #pragma unroll
      for (int s = 0; s < 16; s++)
        bfm[ni][s] = Bs[buf][(s >> 3) * 16 + rhalf * 8 + (s & 7)][c + rlow];
    }
    #pragma unroll
    for (int mi = 0; mi < 2; mi++)
      #pragma unroll
      for (int ni = 0; ni < 4; ni++)
        acc[mi][ni] = wmma_bf16(af[mi], bfm[ni], acc[mi][ni]);
    __syncthreads();   /* buffer consumed; safe to re-issue next round */
  }

  #pragma unroll
  for (int ni = 0; ni < 4; ni++) {
    int col = n0 + wn + ni * 16 + rlow;
    float bv = bias[col];
    #pragma unroll
    for (int mi = 0; mi < 2; mi++) {
      #pragma unroll
      for (int j = 0; j < 8; j++) {
        int row = m0 + wm + mi * 16 + rhalf * 8 + j;
        C[(size_t)row * N + col] = acc[mi][ni][j] + bv;
      }
    }
  }
}

/* ---- per-head l2 norm + layout change, fp32 -> bf16.
   mode 0: Q  -> [B,H,Lq,dh], scaled by QK_SCALE
   mode 1: K  -> [B,H,dh,Lk]  (pre-transposed for QK^T B-fragments)
   mode 2: V  -> [B,H,Lk,dh], no norm
   One wave per (b,l,h); 2 elements per lane.                      */
__global__ __launch_bounds__(256) void norm_kernel(const float* __restrict__ X,
                                                   bf16_t* __restrict__ O, int mode)
{
  int g = blockIdx.x * 8 + (threadIdx.x >> 5);
  int lane = threadIdx.x & 31;
  int h = g & (NH - 1);
  int row = g >> 4;            /* b*LQ + l */
  int b = row >> 11;
  int l = row & (LQ - 1);
  const float2 xy = *reinterpret_cast<const float2*>(
      &X[(size_t)row * DM + h * DH + lane * 2]);
  float inv = 1.f;
  if (mode != 2) {
    float s = xy.x * xy.x + xy.y * xy.y;
    #pragma unroll
    for (int off = 16; off >= 1; off >>= 1) s += __shfl_xor(s, off, 32);
    inv = 1.f / fmaxf(sqrtf(s), 1e-12f);
    if (mode == 0) inv *= QK_SCALE;
  }
  bf16_t a = (bf16_t)(xy.x * inv), c = (bf16_t)(xy.y * inv);
  size_t base = (size_t)(b * NH + h);
  if (mode == 1) {
    bf16_t* p = O + (base * DH + lane * 2) * LK + l;
    p[0] = a; p[LK] = c;
  } else {
    bf16_t* p = O + (base * LQ + l) * DH + lane * 2;
    p[0] = a; p[1] = c;
  }
}

/* ---- vbar[b,h,d] = mean over valid keys of projected V ---- */
__global__ __launch_bounds__(64) void vbar_kernel(const float* __restrict__ Vf,
    const int* __restrict__ mask, float* __restrict__ vbar)
{
  int bh = blockIdx.x; int b = bh >> 4, h = bh & 15;
  int d = threadIdx.x;
  float s = 0.f; int cnt = 0;
  for (int k = 0; k < LK; k++) {
    if (mask[b * LK + k] == 0) {
      s += Vf[((size_t)(b * LK + k)) * DM + h * DH + d];
      cnt++;
    }
  }
  vbar[bh * DH + d] = s / (float)(cnt > 0 ? cnt : 1);
}

/* ---- fused flash attention, 8 waves x 16 queries per WG, 128-key tiles.
   K/V tiles staged with async global->LDS copies (ASYNCcnt).       ---- */
__global__ __launch_bounds__(256) void attn_kernel(const bf16_t* __restrict__ Qn,
    const bf16_t* __restrict__ KT, const bf16_t* __restrict__ Vh,
    const int* __restrict__ mask, const float* __restrict__ vbar,
    bf16_t* __restrict__ Out)
{
  __shared__ bf16_t Ks[64][136];      /* KT tile: [d][key]   */
  __shared__ bf16_t Vs[128][72];      /* V tile:  [key][d]   */
  __shared__ bf16_t Ps[8][16][40];    /* per-wave P re-layout pad */
  const int tid = threadIdx.x, lane = tid & 31, wid = tid >> 5;
  const int rhalf = lane >> 4, rlow = lane & 15;
  const int b = blockIdx.z, h = blockIdx.y;
  const int q0 = blockIdx.x * 128 + wid * 16;
  const size_t headQ = (size_t)(b * NH + h) * LQ;
  const size_t headK = (size_t)(b * NH + h) * DH;

  /* Q fragments for d-chunks 0 and 32 (dh = 64) */
  v16bf aq[2];
  #pragma unroll
  for (int c = 0; c < 2; c++) {
    const bf16_t* qp = Qn + (headQ + q0 + rlow) * DH + c * 32 + rhalf * 8;
    aq[c] = mk16(*reinterpret_cast<const v8bf*>(qp),
                 *reinterpret_cast<const v8bf*>(qp + 16));
  }
  float m[8], lsum[8];
  #pragma unroll
  for (int j = 0; j < 8; j++) { m[j] = -1e30f; lsum[j] = 0.f; }
  v8f accO[4];
  #pragma unroll
  for (int ni = 0; ni < 4; ni++) accO[ni] = 0.f;

  for (int kt = 0; kt < LK / 128; kt++) {
    /* async stage KT tile 64x128 (16 chunks/row) */
    #pragma unroll
    for (int i = 0; i < 4; i++) {
      int idx8 = tid + i * 256;
      int r = idx8 >> 4, c8 = (idx8 & 15) * 8;
      async_b128(LDSOFF(&Ks[r][c8]), KT,
                 (unsigned)(((headK + r) * LK + kt * 128 + c8) * 2));
    }
    /* async stage V tile 128x64 (8 chunks/row) */
    #pragma unroll
    for (int i = 0; i < 4; i++) {
      int idx8 = tid + i * 256;
      int r = idx8 >> 3, c8 = (idx8 & 7) * 8;
      async_b128(LDSOFF(&Vs[r][c8]), Vh,
                 (unsigned)(((headQ + kt * 128 + r) * DH + c8) * 2));
    }
    asm volatile("s_wait_asynccnt 0x0" ::: "memory");
    __syncthreads();

    int mk[8];
    #pragma unroll
    for (int t = 0; t < 8; t++) mk[t] = mask[b * LK + kt * 128 + t * 16 + rlow];

    v8f s[8];
    #pragma unroll
    for (int t = 0; t < 8; t++) s[t] = 0.f;
    #pragma unroll
    for (int t = 0; t < 8; t++)
      #pragma unroll
      for (int c = 0; c < 2; c++) {
        v16bf bk;
        #pragma unroll
        for (int s = 0; s < 16; s++)
          bk[s] = Ks[c * 32 + (s >> 3) * 16 + rhalf * 8 + (s & 7)][t * 16 + rlow];
        s[t] = wmma_bf16(aq[c], bk, s[t]);
      }

    /* online softmax; row j (j + 8*rhalf) lives in accumulator element j */
    #pragma unroll
    for (int j = 0; j < 8; j++) {
      float mx = -1e30f;
      #pragma unroll
      for (int t = 0; t < 8; t++) mx = fmaxf(mx, mk[t] ? -1e30f : s[t][j]);
      #pragma unroll
      for (int off = 8; off >= 1; off >>= 1) mx = fmaxf(mx, __shfl_xor(mx, off, 32));
      float mn = fmaxf(m[j], mx);
      float alpha = __expf(m[j] - mn);
      float rs = 0.f;
      #pragma unroll
      for (int t = 0; t < 8; t++) {
        float p = mk[t] ? 0.f : __expf(s[t][j] - mn);
        s[t][j] = p; rs += p;
      }
      #pragma unroll
      for (int off = 8; off >= 1; off >>= 1) rs += __shfl_xor(rs, off, 32);
      lsum[j] = lsum[j] * alpha + rs;
      m[j] = mn;
      #pragma unroll
      for (int ni = 0; ni < 4; ni++) accO[ni][j] *= alpha;
    }

    /* P @ V in 32-key chunks: C-layout -> LDS -> A-layout -> WMMA */
    #pragma unroll
    for (int c = 0; c < 4; c++) {
      #pragma unroll
      for (int tt = 0; tt < 2; tt++) {
        int t = c * 2 + tt;
        #pragma unroll
        for (int j = 0; j < 8; j++)
          Ps[wid][rhalf * 8 + j][tt * 16 + rlow] = (bf16_t)s[t][j];
      }
      v16bf pa = mk16(*reinterpret_cast<v8bf*>(&Ps[wid][rlow][rhalf * 8]),
                      *reinterpret_cast<v8bf*>(&Ps[wid][rlow][rhalf * 8 + 16]));
      #pragma unroll
      for (int ni = 0; ni < 4; ni++) {
        v16bf bv;
        #pragma unroll
        for (int s = 0; s < 16; s++)
          bv[s] = Vs[c * 32 + (s >> 3) * 16 + rhalf * 8 + (s & 7)][ni * 16 + rlow];
        accO[ni] = wmma_bf16(pa, bv, accO[ni]);
      }
    }
    __syncthreads();
  }

  #pragma unroll
  for (int ni = 0; ni < 4; ni++) {
    int d = ni * 16 + rlow;
    float vb = vbar[(b * NH + h) * DH + d];
    #pragma unroll
    for (int j = 0; j < 8; j++) {
      int q = q0 + rhalf * 8 + j;
      float o = (1.f - EPS_SM) * accO[ni][j] / fmaxf(lsum[j], 1e-8f) + EPS_SM * vb;
      Out[((size_t)(b * LQ + q)) * DM + h * DH + d] = (bf16_t)o;
    }
  }
}

extern "C" void kernel_launch(void* const* d_in, const int* in_sizes, int n_in,
                              void* d_out, int out_size, void* d_ws, size_t ws_size,
                              hipStream_t stream)
{
  (void)in_sizes; (void)n_in; (void)out_size; (void)ws_size;
  const float* q_in = (const float*)d_in[0];
  const float* k_in = (const float*)d_in[1];
  const float* v_in = (const float*)d_in[2];
  const int*   mask = (const int*)d_in[3];
  const float* Wq = (const float*)d_in[4];  const float* bq = (const float*)d_in[5];
  const float* Wk = (const float*)d_in[6];  const float* bk = (const float*)d_in[7];
  const float* Wv = (const float*)d_in[8];  const float* bv = (const float*)d_in[9];
  const float* Wo = (const float*)d_in[10]; const float* bo = (const float*)d_in[11];
  float* out = (float*)d_out;

  const int M = B_BATCH * LQ;                 /* 4096 */
  char* ws = (char*)d_ws;
  size_t off = 0;
  auto alloc = [&](size_t bytes) -> char* {
    char* p = ws + off; off += (bytes + 255) & ~(size_t)255; return p;
  };
  bf16_t* Xq  = (bf16_t*)alloc((size_t)M * DM * 2);
  bf16_t* Xk  = (bf16_t*)alloc((size_t)M * DM * 2);
  bf16_t* Xv  = (bf16_t*)alloc((size_t)M * DM * 2);
  bf16_t* WTq = (bf16_t*)alloc((size_t)DM * DM * 2);
  bf16_t* WTk = (bf16_t*)alloc((size_t)DM * DM * 2);
  bf16_t* WTv = (bf16_t*)alloc((size_t)DM * DM * 2);
  bf16_t* WTo = (bf16_t*)alloc((size_t)DM * DM * 2);
  float*  Qf  = (float*) alloc((size_t)M * DM * 4);
  float*  Kf  = (float*) alloc((size_t)M * DM * 4);
  float*  Vf  = (float*) alloc((size_t)M * DM * 4);
  bf16_t* Qn  = (bf16_t*)alloc((size_t)M * DM * 2);
  bf16_t* KTn = (bf16_t*)alloc((size_t)M * DM * 2);
  bf16_t* Vhn = (bf16_t*)alloc((size_t)M * DM * 2);
  float*  vbp = (float*) alloc((size_t)B_BATCH * NH * DH * 4);
  bf16_t* aOb = (bf16_t*)alloc((size_t)M * DM * 2);

  const dim3 blk(256);
  const int cvGrid = M * DM / 1024;
  conv_kernel<<<cvGrid, blk, 0, stream>>>(q_in, Xq);
  conv_kernel<<<cvGrid, blk, 0, stream>>>(k_in, Xk);
  conv_kernel<<<cvGrid, blk, 0, stream>>>(v_in, Xv);

  const int wtGrid = DM * DM / 256;
  wt_kernel<<<wtGrid, blk, 0, stream>>>(Wq, WTq);
  wt_kernel<<<wtGrid, blk, 0, stream>>>(Wk, WTk);
  wt_kernel<<<wtGrid, blk, 0, stream>>>(Wv, WTv);
  wt_kernel<<<wtGrid, blk, 0, stream>>>(Wo, WTo);

  const dim3 gGrid(DM / 128, M / 128);
  gemm_kernel<<<gGrid, blk, 0, stream>>>(Xq, WTq, bq, Qf, M, DM, DM);
  gemm_kernel<<<gGrid, blk, 0, stream>>>(Xk, WTk, bk, Kf, M, DM, DM);
  gemm_kernel<<<gGrid, blk, 0, stream>>>(Xv, WTv, bv, Vf, M, DM, DM);

  const int nGrid = M * NH / 8;
  norm_kernel<<<nGrid, blk, 0, stream>>>(Qf, Qn, 0);
  norm_kernel<<<nGrid, blk, 0, stream>>>(Kf, KTn, 1);
  norm_kernel<<<nGrid, blk, 0, stream>>>(Vf, Vhn, 2);

  vbar_kernel<<<B_BATCH * NH, dim3(64), 0, stream>>>(Vf, mask, vbp);

  attn_kernel<<<dim3(LQ / 128, NH, B_BATCH), blk, 0, stream>>>(
      Qn, KTn, Vhn, mask, vbp, aOb);

  gemm_kernel<<<gGrid, blk, 0, stream>>>(aOb, WTo, bo, out, M, DM, DM);
}
